// ContrastiveLoss_28630251995675
// MI455X (gfx1250) — matrix operand order repaired and checked by reading, not compiled, not run on hardware
//
#include <hip/hip_runtime.h>
#include <hip/hip_bf16.h>
#include <math.h>

// ---------------------------------------------------------------------------
// Problem constants (from reference)
// ---------------------------------------------------------------------------
static constexpr int BN   = 16384;  // batch
static constexpr int PP   = 128;    // projection dim
static constexpr int DIMG = 2048;
static constexpr int DNUM = 256;

typedef _Float16 v16h __attribute__((ext_vector_type(16)));
typedef _Float16 v8h  __attribute__((ext_vector_type(8)));
typedef float    v8f  __attribute__((ext_vector_type(8)));
typedef float    v4f  __attribute__((ext_vector_type(4)));

// ---------------------------------------------------------------------------
// WMMA helper:  D(16x16 f32) = A(16x32 f16) * B(32x16 f16) + C
// ---------------------------------------------------------------------------
static __device__ inline v8f wmma_f16(v16h a, v16h b, v8f c) {
    return __builtin_amdgcn_wmma_f32_16x16x32_f16(
        /*neg_a=*/false, a, /*neg_b=*/false, b,
        /*c_mod=*/(short)0, c, /*reuse_a=*/false, /*reuse_b=*/false);
}

static __device__ inline v16h cat8(v8h lo, v8h hi8) {
    union { v16h v; v8h h[2]; } u;
    u.h[0] = lo;
    u.h[1] = hi8;
    return u.v;
}

// A-matrix 16x32 f16 layout (ISA 7.12.2):
//  lanes 0-15 : row = lane,    elems K = k0+0..7  and K = k0+16..23
//  lanes 16-31: row = lane-16, elems K = k0+8..15 and K = k0+24..31
// -> per lane: base k = k0 + hi*8, chunks [base, base+8) and [base+16, base+24)
// Each chunk = 16 contiguous, 16B-aligned bytes -> one b128 load.
static __device__ inline v16h load_a_f16(const _Float16* X, int ld, int row,
                                         int k0, int hi) {
    const _Float16* p = X + (size_t)row * ld + k0 + hi * 8;
    return cat8(*(const v8h*)p, *(const v8h*)(p + 16));
}

// f32 source variant (two 32B chunks, float4 loads + cvt to f16)
static __device__ inline v16h load_a_f32(const float* X, int ld, int row,
                                         int k0, int hi) {
    const float* p = X + (size_t)row * ld + k0 + hi * 8;
    v4f q0 = *(const v4f*)(p);
    v4f q1 = *(const v4f*)(p + 4);
    v4f q2 = *(const v4f*)(p + 16);
    v4f q3 = *(const v4f*)(p + 20);
    v16h a;
#pragma unroll
    for (int i = 0; i < 4; ++i) {
        a[i]      = (_Float16)q0[i];
        a[4 + i]  = (_Float16)q1[i];
        a[8 + i]  = (_Float16)q2[i];
        a[12 + i] = (_Float16)q3[i];
    }
    return a;
}

// B-matrix 32x16 f16 layout (same K order as A):
//  lanes 0-15 : col = lane,    elems K = k0+0..7  and K = k0+16..23
//  lanes 16-31: col = lane-16, elems K = k0+8..15 and K = k0+24..31
// Source stored K-contiguous per column (W^T row-major, leading dim ldk):
// per lane two 16B chunks -> two b128 loads.
static __device__ inline v16h load_bt_f16(const _Float16* WT, int ldk, int col,
                                          int k0, int hi) {
    const _Float16* p = WT + (size_t)col * ldk + k0 + hi * 8;
    return cat8(*(const v8h*)p, *(const v8h*)(p + 16));
}

// ---------------------------------------------------------------------------
// Small utility kernels
// ---------------------------------------------------------------------------
// dst = f16(transpose(src)) : src f32 [Dk][P] row-major -> dst f16 [P][Dk]
__global__ void k_transpose_f16(const float* __restrict__ s,
                                _Float16* __restrict__ d, int Dk) {
    int i = blockIdx.x * 256 + threadIdx.x;
    if (i < Dk * PP) {
        int k = i / PP;
        int n = i - k * PP;
        d[(size_t)n * Dk + k] = (_Float16)s[i];
    }
}

__global__ void k_zero(float* __restrict__ a, int n) {
    int i = blockIdx.x * 256 + threadIdx.x;
    if (i < n) a[i] = 0.0f;
}

// ---------------------------------------------------------------------------
// Fused MLP + L2-normalize.
// Block = 256 threads (8 waves), computes 16 rows x 128 cols.
// Wave w owns col-tile [w*16, w*16+16).
// GEMM1: h = relu(X @ W1 + b1)  (K = Dk), h -> LDS as f16
// GEMM2: o = h @ W2 + b2        (K = 128)
// Then row-wise L2 norm, write f32 + f16 embeddings.
// Weights supplied pre-transposed (W^T row-major f16).
// ---------------------------------------------------------------------------
__global__ __launch_bounds__(256) void k_mlp_norm(
    const float* __restrict__ X, const _Float16* __restrict__ W1T,
    const float* __restrict__ b1, const _Float16* __restrict__ W2T,
    const float* __restrict__ b2, float* __restrict__ outf,
    _Float16* __restrict__ outh, int Dk) {
    __shared__ __align__(16) _Float16 hbuf[16 * PP];
    __shared__ float rowsq[16];

    const int tid  = threadIdx.x;
    const int w    = tid >> 5;
    const int lane = tid & 31;
    const int lh   = lane & 15;
    const int hi   = lane >> 4;
    const int row0 = blockIdx.x * 16;
    const int col  = w * 16 + lh;

    if (tid < 16) rowsq[tid] = 0.0f;

    // ---- GEMM1 ----
    v8f acc = {0.f, 0.f, 0.f, 0.f, 0.f, 0.f, 0.f, 0.f};
    const int arow = row0 + lh;
    for (int k0 = 0; k0 < Dk; k0 += 32) {
        v16h a = load_a_f32(X, Dk, arow, k0, hi);
        v16h b = load_bt_f16(W1T, Dk, col, k0, hi);
        acc = wmma_f16(a, b, acc);
    }
    {
        float bv = b1[col];
#pragma unroll
        for (int j = 0; j < 8; ++j) {
            float h = acc[j] + bv;
            h = h > 0.0f ? h : 0.0f;
            hbuf[(size_t)(j + hi * 8) * PP + col] = (_Float16)h;
        }
    }
    __syncthreads();

    // ---- GEMM2 (A from LDS, b128 ds loads) ----
    v8f acc2 = {0.f, 0.f, 0.f, 0.f, 0.f, 0.f, 0.f, 0.f};
#pragma unroll
    for (int k0 = 0; k0 < PP; k0 += 32) {
        v16h a = load_a_f16(hbuf, PP, lh, k0, hi);
        v16h b = load_bt_f16(W2T, PP, col, k0, hi);
        acc2 = wmma_f16(a, b, acc2);
    }
    float o[8];
    {
        float bv = b2[col];
#pragma unroll
        for (int j = 0; j < 8; ++j) o[j] = acc2[j] + bv;
    }

    // ---- row sums of squares ----
#pragma unroll
    for (int j = 0; j < 8; ++j) {
        float v = o[j] * o[j];
        v += __shfl_xor(v, 1, 32);
        v += __shfl_xor(v, 2, 32);
        v += __shfl_xor(v, 4, 32);
        v += __shfl_xor(v, 8, 32);
        if (lh == 0) atomicAdd(&rowsq[j + hi * 8], v);
    }
    __syncthreads();

    // ---- normalize + store ----
#pragma unroll
    for (int j = 0; j < 8; ++j) {
        int r = j + hi * 8;
        float nrm = sqrtf(rowsq[r]);
        float s = 1.0f / fmaxf(nrm, 1e-12f);
        float val = o[j] * s;
        size_t idx = (size_t)(row0 + r) * PP + col;
        outf[idx] = val;
        outh[idx] = (_Float16)val;
    }
}

// ---------------------------------------------------------------------------
// diag[i] = dot(ip[i], np[i]) / temp   (f32, 1 wave per row)
// ---------------------------------------------------------------------------
__global__ __launch_bounds__(256) void k_diag(const float* __restrict__ IPf,
                                              const float* __restrict__ NPf,
                                              const float* __restrict__ lt,
                                              float* __restrict__ diag) {
    const int w = threadIdx.x >> 5;
    const int lane = threadIdx.x & 31;
    const int row = blockIdx.x * 8 + w;
    const float* a = IPf + (size_t)row * PP;
    const float* b = NPf + (size_t)row * PP;
    float s = 0.0f;
#pragma unroll
    for (int i = 0; i < PP / 32; ++i) s += a[lane + i * 32] * b[lane + i * 32];
    s += __shfl_xor(s, 1, 32);
    s += __shfl_xor(s, 2, 32);
    s += __shfl_xor(s, 4, 32);
    s += __shfl_xor(s, 8, 32);
    s += __shfl_xor(s, 16, 32);
    if (lane == 0) diag[row] = s * __expf(-lt[0]);
}

// ---------------------------------------------------------------------------
// Fused logits tile + shifted exp-sum accumulation.
// Grid (BN/128, BN/128), block 256 = 8 waves. Wave w owns rows
// [by*128 + w*16, +16) over cols [bx*128, +128). SHIFT = 1/temp = max logit.
// ---------------------------------------------------------------------------
__global__ __launch_bounds__(256) void k_logits(
    const _Float16* __restrict__ IP, const _Float16* __restrict__ NP,
    const float* __restrict__ lt, float* __restrict__ row_sum,
    float* __restrict__ col_sum) {
    __shared__ float cls[PP];

    const int tid  = threadIdx.x;
    const int w    = tid >> 5;
    const int lane = tid & 31;
    const int lh   = lane & 15;
    const int hi   = lane >> 4;

    if (tid < PP) cls[tid] = 0.0f;
    __syncthreads();

    const float inv_temp = __expf(-lt[0]);  // also the lse shift
    const int r0 = blockIdx.y * 128 + w * 16;
    const int c0 = blockIdx.x * 128;

    // A strip: 16 rows x K=128, kept in registers (4 x v16h = 32 VGPRs)
    v16h a[4];
    const int arow = r0 + lh;
#pragma unroll
    for (int ks = 0; ks < 4; ++ks) a[ks] = load_a_f16(IP, PP, arow, ks * 32, hi);

    float rowacc[8] = {0.f, 0.f, 0.f, 0.f, 0.f, 0.f, 0.f, 0.f};

#pragma unroll
    for (int ct = 0; ct < 8; ++ct) {
        const int col = c0 + ct * 16 + lh;
        v8f acc = {0.f, 0.f, 0.f, 0.f, 0.f, 0.f, 0.f, 0.f};
#pragma unroll
        for (int ks = 0; ks < 4; ++ks) {
            v16h b = load_bt_f16(NP, PP, col, ks * 32, hi);
            acc = wmma_f16(a[ks], b, acc);
        }
        float cacc = 0.0f;
#pragma unroll
        for (int j = 0; j < 8; ++j) {
            float lg = acc[j] * inv_temp;
            float e = __expf(lg - inv_temp);  // exp(logit - SHIFT), <= 1
            rowacc[j] += e;
            cacc += e;
        }
        cacc += __shfl_xor(cacc, 16, 32);            // fold row-halves
        if (hi == 0) atomicAdd(&cls[ct * 16 + lh], cacc);
    }

    // row partials -> global
#pragma unroll
    for (int j = 0; j < 8; ++j) {
        float r = rowacc[j];
        r += __shfl_xor(r, 1, 32);
        r += __shfl_xor(r, 2, 32);
        r += __shfl_xor(r, 4, 32);
        r += __shfl_xor(r, 8, 32);
        if (lh == 0) atomicAdd(&row_sum[r0 + j + hi * 8], r);
    }

    __syncthreads();
    if (tid < PP) atomicAdd(&col_sum[c0 + tid], cls[tid]);
}

// ---------------------------------------------------------------------------
// loss = SHIFT + sum_i( log(rs_i) + log(cs_i) - 2*diag_i ) / (2B)
// ---------------------------------------------------------------------------
__global__ __launch_bounds__(256) void k_final(const float* __restrict__ rs,
                                               const float* __restrict__ cs,
                                               const float* __restrict__ dg,
                                               const float* __restrict__ lt,
                                               float* __restrict__ out) {
    __shared__ float red[256];
    float s = 0.0f;
    for (int i = threadIdx.x; i < BN; i += 256)
        s += logf(rs[i]) + logf(cs[i]) - 2.0f * dg[i];
    red[threadIdx.x] = s;
    __syncthreads();
    for (int off = 128; off > 0; off >>= 1) {
        if (threadIdx.x < off) red[threadIdx.x] += red[threadIdx.x + off];
        __syncthreads();
    }
    if (threadIdx.x == 0)
        out[0] = __expf(-lt[0]) + red[0] / (2.0f * (float)BN);
}

// ---------------------------------------------------------------------------
// Launcher
// ---------------------------------------------------------------------------
extern "C" void kernel_launch(void* const* d_in, const int* in_sizes, int n_in,
                              void* d_out, int out_size, void* d_ws,
                              size_t ws_size, hipStream_t stream) {
    (void)in_sizes; (void)n_in; (void)out_size; (void)ws_size;

    const float* img = (const float*)d_in[0];
    const float* num = (const float*)d_in[1];
    const float* Wi1 = (const float*)d_in[2];
    const float* bi1 = (const float*)d_in[3];
    const float* Wi2 = (const float*)d_in[4];
    const float* bi2 = (const float*)d_in[5];
    const float* Wn1 = (const float*)d_in[6];
    const float* bn1 = (const float*)d_in[7];
    const float* Wn2 = (const float*)d_in[8];
    const float* bn2 = (const float*)d_in[9];
    const float* lt  = (const float*)d_in[10];

    // workspace carving (256B aligned)
    char* base = (char*)d_ws;
    size_t off = 0;
    auto carve = [&](size_t bytes) {
        char* p = base + off;
        off += (bytes + 255) & ~(size_t)255;
        return p;
    };
    _Float16* Wi1t = (_Float16*)carve((size_t)DIMG * PP * 2);  // [P][DIMG]
    _Float16* Wn1t = (_Float16*)carve((size_t)DNUM * PP * 2);  // [P][DNUM]
    _Float16* Wi2t = (_Float16*)carve((size_t)PP * PP * 2);    // [P][P]
    _Float16* Wn2t = (_Float16*)carve((size_t)PP * PP * 2);    // [P][P]
    float*    ipf  = (float*)carve((size_t)BN * PP * 4);
    float*    npf  = (float*)carve((size_t)BN * PP * 4);
    _Float16* iph  = (_Float16*)carve((size_t)BN * PP * 2);
    _Float16* nph  = (_Float16*)carve((size_t)BN * PP * 2);
    float*    sums = (float*)carve((size_t)2 * BN * 4);  // row_sum | col_sum
    float*    dgv  = (float*)carve((size_t)BN * 4);
    float* row_sum = sums;
    float* col_sum = sums + BN;

    // 1) weight transpose+convert, accumulator zeroing
    k_transpose_f16<<<(DIMG * PP + 255) / 256, 256, 0, stream>>>(Wi1, Wi1t, DIMG);
    k_transpose_f16<<<(DNUM * PP + 255) / 256, 256, 0, stream>>>(Wn1, Wn1t, DNUM);
    k_transpose_f16<<<(PP * PP + 255) / 256, 256, 0, stream>>>(Wi2, Wi2t, PP);
    k_transpose_f16<<<(PP * PP + 255) / 256, 256, 0, stream>>>(Wn2, Wn2t, PP);
    k_zero<<<(2 * BN + 255) / 256, 256, 0, stream>>>(sums, 2 * BN);

    // 2) fused MLP + normalize per modality
    k_mlp_norm<<<BN / 16, 256, 0, stream>>>(img, Wi1t, bi1, Wi2t, bi2, ipf, iph, DIMG);
    k_mlp_norm<<<BN / 16, 256, 0, stream>>>(num, Wn1t, bn1, Wn2t, bn2, npf, nph, DNUM);

    // 3) diagonal
    k_diag<<<BN / 8, 256, 0, stream>>>(ipf, npf, lt, dgv);

    // 4) fused logits + shifted exp-sum accumulation
    dim3 grid(BN / 128, BN / 128);
    k_logits<<<grid, 256, 0, stream>>>(iph, nph, lt, row_sum, col_sum);

    // 5) final reduction
    k_final<<<1, 256, 0, stream>>>(row_sum, col_sum, dgv, lt, (float*)d_out);
}
